// KmeansEncoder_77386720739407
// MI455X (gfx1250) — matrix operand order repaired and checked by reading, not compiled, not run on hardware
//
#include <hip/hip_runtime.h>

// ---- problem constants (from reference) ----
#define BATCH 4096
#define TT    16
#define KDIM  1152      // T*N*D = 16*24*3
#define CC    512
#define NFULL (CC * 16) // 8192 im2col'd weight rows: n = c*16 + t (t=15 dummy)

typedef __attribute__((ext_vector_type(16))) _Float16 v16h;
typedef __attribute__((ext_vector_type(8)))  _Float16 v8h;
typedef __attribute__((ext_vector_type(4)))  _Float16 v4h;
typedef __attribute__((ext_vector_type(8)))  float    v8f;
typedef __attribute__((ext_vector_type(4)))  float    v4f;

// ---------------------------------------------------------------------------
// Kernel 1a: pack x (f32 [4096][1152]) -> f16 [4096][1152]
// ---------------------------------------------------------------------------
__global__ __launch_bounds__(256) void pack_x_f16(const float* __restrict__ x,
                                                  _Float16* __restrict__ xh) {
  size_t i = ((size_t)blockIdx.x * 256 + threadIdx.x) * 4;
  v4f f = *(const v4f*)(x + i);
  v4h h;
#pragma unroll
  for (int e = 0; e < 4; ++e) h[e] = (_Float16)f[e];
  *(v4h*)(xh + i) = h;
}

// ---------------------------------------------------------------------------
// Kernel 1b: im2col'd, time-shifted, zero-padded weights in f16:
//   Wh[(c*16+t)][k] = W[c][k + (7-t)*72]  if t<15 and 0 <= k+(7-t)*72 < 1152
//                   = 0                   otherwise (incl. the t=15 dummy col)
// Validity boundaries are multiples of 72; 4-aligned 4-chunks are wholly
// valid or invalid.
// ---------------------------------------------------------------------------
__global__ __launch_bounds__(256) void pack_w_f16(const float* __restrict__ W,
                                                  _Float16* __restrict__ wh) {
  const int gid = blockIdx.x * 256 + threadIdx.x;   // [0, 8192*288)
  const int n   = gid / (KDIM / 4);
  const int kc  = (gid % (KDIM / 4)) * 4;
  const int c   = n >> 4;
  const int t   = n & 15;
  const int ks  = kc + (7 - t) * 72;
  const bool ok = (t < 15) && (ks >= 0) && (ks <= KDIM - 4);
  v4f f = (v4f){0.f, 0.f, 0.f, 0.f};
  if (ok) f = *(const v4f*)(W + (size_t)c * KDIM + ks);
  v4h h;
#pragma unroll
  for (int e = 0; e < 4; ++e) h[e] = (_Float16)f[e];
  *(v4h*)(wh + (size_t)n * KDIM + kc) = h;
}

// ---------------------------------------------------------------------------
// Kernel 2: WMMA GEMM logits'[b, (c,t)] = Xh[b,:] . Wh[(c,t),:] with fused
// max-over-t + bias.  Each 16x16 D tile = 16 rows x one code's 16 t-columns.
// Block: 8 waves, tile 128 rows x 16 codes; wave: 4 Mtiles x 4 codes
// -> 16 v_wmma_f32_16x16x32_f16 per K-step, 36 K-steps.
// Epilogue: per-wave LDS transpose of two tiles at a time; lanes 0-15 reduce
// tile j, lanes 16-31 reduce tile j+1; one full-wave scatter store per pair.
// ---------------------------------------------------------------------------
#define TSTRIDE 20            // floats per LDS tile row (80 B: 16B-aligned, bank-spread)
#define TILE2   352           // float offset of second tile in a wave's patch
#define WPATCH  704           // floats per wave patch

__global__ __launch_bounds__(256) void conv_gemm_max(
    const _Float16* __restrict__ Xh,      // [4096][1152] f16
    const _Float16* __restrict__ Wh,      // [8192][1152] f16
    const float*    __restrict__ bias,    // [512]
    float*          __restrict__ logits)  // [4096][512] f32
{
  __shared__ float lds[8 * WPATCH];       // 22.5 KB of 320 KB

  const int lane = threadIdx.x & 31;
  const int wave = threadIdx.x >> 5;
  const int wm   = wave & 1;   // row half of block tile (64 rows each)
  const int wn   = wave >> 1;  // code quad 0..3
  const int lrow = lane & 15;
  const int half = lane >> 4;  // K-split half of the wave

  const int rowBlock  = blockIdx.x * 128;
  const int codeBlock = blockIdx.y * 16;

  const int row0  = rowBlock + wm * 64 + lrow;          // A row, subtile 0
  const int wrow0 = (codeBlock + wn * 4) * 16 + lrow;   // Wh row, code tile 0

  v8f acc[4][4];
#pragma unroll
  for (int i = 0; i < 4; ++i)
#pragma unroll
    for (int j = 0; j < 4; ++j)
      acc[i][j] = (v8f){0.f,0.f,0.f,0.f,0.f,0.f,0.f,0.f};

  for (int kk = 0; kk < KDIM; kk += 32) {
    // A: lanes<16 hold K = kk+0..7 & kk+16..23, lanes>=16 hold +8 of each
    v16h a[4];
    {
      const int kbase = kk + half * 8;
#pragma unroll
      for (int i = 0; i < 4; ++i) {
        const _Float16* p = Xh + (size_t)(row0 + i * 16) * KDIM + kbase;
        v8h lo = *(const v8h*)(p);
        v8h hi = *(const v8h*)(p + 16);
        a[i] = __builtin_shufflevector(lo, hi, 0,1,2,3,4,5,6,7,8,9,10,11,12,13,14,15);
      }
    }
    // B: lanes<16 hold K = kk+0..7 & kk+16..23 (col t=lrow), lanes>=16 hold +8 of each
    v16h bf[4];
    {
      const int kbase = kk + half * 8;
#pragma unroll
      for (int j = 0; j < 4; ++j) {
        const _Float16* p = Wh + (size_t)(wrow0 + j * 16) * KDIM + kbase;
        v8h lo = *(const v8h*)(p);
        v8h hi = *(const v8h*)(p + 16);
        bf[j] = __builtin_shufflevector(lo, hi, 0,1,2,3,4,5,6,7,8,9,10,11,12,13,14,15);
      }
    }
#pragma unroll
    for (int i = 0; i < 4; ++i)
#pragma unroll
      for (int j = 0; j < 4; ++j)
        acc[i][j] = __builtin_amdgcn_wmma_f32_16x16x32_f16(
            false, a[i], false, bf[j], (short)0, acc[i][j], false, false);
  }

  // ---- Epilogue via per-wave LDS transpose ----
  // D element v of lane L = (M = v + 8*half, N = lrow = t).
  float* wbase = lds + wave * WPATCH;
#pragma unroll
  for (int i = 0; i < 4; ++i) {
#pragma unroll
    for (int jp = 0; jp < 4; jp += 2) {
      // scatter the two tiles into LDS: elem (M,N) at [jj*TILE2 + M*TSTRIDE + N]
#pragma unroll
      for (int jj = 0; jj < 2; ++jj) {
        float* tb = wbase + jj * TILE2 + (half * 8) * TSTRIDE + lrow;
#pragma unroll
        for (int v = 0; v < 8; ++v) tb[v * TSTRIDE] = acc[i][jp + jj][v];
      }
      // lanes 0-15 reduce tile jp, lanes 16-31 reduce tile jp+1; each lane
      // owns row M = lrow: contiguous 15 valid t's (t=15 dummy excluded).
      const float* rp = wbase + half * TILE2 + lrow * TSTRIDE;
      float mx = rp[0];
#pragma unroll
      for (int e = 1; e < 15; ++e) mx = fmaxf(mx, rp[e]);

      const int c   = codeBlock + wn * 4 + jp + half;
      const int row = rowBlock + wm * 64 + i * 16 + lrow;
      logits[(size_t)row * CC + c] = mx + bias[c];
    }
  }
}

// ---------------------------------------------------------------------------
// Kernel 3: per-sample argmax over 512 codes (first-max tie-break, matching
// jnp.argmax) + gather centers[cls] -> out. One wave per sample.
// ---------------------------------------------------------------------------
__global__ __launch_bounds__(256) void argmax_gather(
    const float* __restrict__ logits,   // [4096][512]
    const float* __restrict__ centers,  // [512][1152]
    float*       __restrict__ out)      // [4096][1152]
{
  const int lane = threadIdx.x & 31;
  const int b    = blockIdx.x * 8 + (threadIdx.x >> 5);

  const float* lr = logits + (size_t)b * CC;
  float best = -__builtin_inff();
  int   bidx = CC;
#pragma unroll
  for (int k = lane; k < CC; k += 32) {
    float v = lr[k];
    if (v > best) { best = v; bidx = k; }   // strided -> per-lane first max
  }
#pragma unroll
  for (int mask = 16; mask; mask >>= 1) {
    float ob = __shfl_xor(best, mask);
    int   oi = __shfl_xor(bidx, mask);
    if (ob > best || (ob == best && oi < bidx)) { best = ob; bidx = oi; }
  }
  const float* src = centers + (size_t)bidx * KDIM;
  float*       dst = out + (size_t)b * KDIM;
#pragma unroll
  for (int k = lane * 4; k < KDIM; k += 128)
    *(v4f*)(dst + k) = *(const v4f*)(src + k);
}

// ---------------------------------------------------------------------------
extern "C" void kernel_launch(void* const* d_in, const int* in_sizes, int n_in,
                              void* d_out, int out_size, void* d_ws, size_t ws_size,
                              hipStream_t stream) {
  (void)in_sizes; (void)n_in; (void)out_size; (void)ws_size;
  const float* x       = (const float*)d_in[0];  // [4096][1][16][24][3]
  const float* W       = (const float*)d_in[1];  // [512][1][16][24][3]
  const float* bias    = (const float*)d_in[2];  // [512]
  const float* centers = (const float*)d_in[3];  // [512][16][24][3]
  float*       out     = (float*)d_out;          // [4096][16][24][3]

  // scratch layout
  char* ws = (char*)d_ws;
  _Float16* Xh     = (_Float16*)ws;                                   //  9.44 MB
  _Float16* Wh     = (_Float16*)(ws + (size_t)BATCH * KDIM * 2);      // 18.87 MB
  float*    logits = (float*)(ws + (size_t)BATCH * KDIM * 2
                                 + (size_t)NFULL * KDIM * 2);         //  8.39 MB

  // 1) packs (independent, both on stream)
  pack_x_f16<<<(BATCH * KDIM) / (256 * 4), 256, 0, stream>>>(x, Xh);
  pack_w_f16<<<(NFULL * (KDIM / 4)) / 256, 256, 0, stream>>>(W, Wh);

  // 2) WMMA conv-GEMM + fused max-over-t + bias
  dim3 grid(BATCH / 128, CC / 16);
  conv_gemm_max<<<grid, 256, 0, stream>>>(Xh, Wh, bias, logits);

  // 3) argmax + codebook gather
  argmax_gather<<<BATCH / 8, 256, 0, stream>>>(logits, centers, out);
}
